// FuseChannelsModuleAtten_20401094656576
// MI455X (gfx1250) — matrix operand order repaired and checked by reading, not compiled, not run on hardware
//
#include <hip/hip_runtime.h>

// ---------------------------------------------------------------- types
typedef __bf16 bf16_t;
typedef __bf16 v16bf __attribute__((ext_vector_type(16)));
typedef float  v8f   __attribute__((ext_vector_type(8)));

struct alignas(16) U4 { unsigned x, y, z, w; };
struct alignas(32) U8 { U4 lo, hi; };

#define BB   4
#define HW   4096   // 64*64 spatial
#define CH   512    // C2/2
#define HID  128    // CH/4

static __device__ __forceinline__ v16bf frag_ld(const void* p0, const void* p1) {
  U8 u;
  u.lo = *(const U4*)p0;
  u.hi = *(const U4*)p1;
  return __builtin_bit_cast(v16bf, u);
}

static __device__ __forceinline__ unsigned short bf16b(float f) {
  bf16_t h = (bf16_t)f;
  return __builtin_bit_cast(unsigned short, h);
}
static __device__ __forceinline__ unsigned pack2(float a, float b) {
  return (unsigned)bf16b(a) | ((unsigned)bf16b(b) << 16);
}

static __device__ __forceinline__ v8f wmma_bf16(v16bf a, v16bf b, v8f c) {
  // D(16x16,f32) = A(16x32,bf16) x B(32x16,bf16) + C
  return __builtin_amdgcn_wmma_f32_16x16x32_bf16(false, a, false, b,
                                                 (short)0, c, false, false);
}

// Async global->LDS copy of one 16B chunk per lane (CDNA5, ASYNCcnt-tracked).
// LDS aperture keeps the byte offset in addr[31:0] (ISA 10.2), so truncating
// the flat pointer yields the LDS address VGPR the instruction expects.
static __device__ __forceinline__ void async_b128(const void* gptr, void* lptr) {
  unsigned lds_off = (unsigned)(size_t)lptr;
  asm volatile("global_load_async_to_lds_b128 %0, %1, off"
               :: "v"(lds_off), "v"(gptr) : "memory");
}
#if __has_builtin(__builtin_amdgcn_s_wait_asynccnt)
#define WAIT_ASYNC(n) __builtin_amdgcn_s_wait_asynccnt(n)
#else
#define WAIT_ASYNC(n) asm volatile("s_wait_asynccnt %0" :: "i"(n) : "memory")
#endif

#if __has_builtin(__builtin_amdgcn_wave_barrier)
#define WAVE_FENCE() __builtin_amdgcn_wave_barrier()
#else
#define WAVE_FENCE() __syncthreads()
#endif

// ================================================================ kernel 0
// Pre-convert both tensors' "orig" halves (V matrices) to bf16 in workspace.
// Layout: Vbf[(tensor*4 + b)*512 + c][p], 33.5 MB total, pure streaming pass.
__global__ __launch_bounds__(256) void vconv_kernel(
    const float* __restrict__ left, const float* __restrict__ right,
    bf16_t* __restrict__ Vbf)
{
  size_t g   = (size_t)blockIdx.x * 256 + threadIdx.x;   // 2,097,152 threads
  size_t e8  = g * 8;
  int    p   = (int)(e8 & (HW - 1));
  int    row = (int)(e8 >> 12);           // (tensor*4+b)*512 + c, 4096 rows
  int    tb  = row >> 9;                  // tensor*4 + b
  int    c   = row & (CH - 1);
  const float* src = ((tb >> 2) ? right : left)
                   + ((size_t)(tb & 3) * (2 * CH) + c) * HW + p;
  float4 a = *(const float4*)(src);
  float4 b = *(const float4*)(src + 4);
  U4 u;
  u.x = pack2(a.x, a.y); u.y = pack2(a.z, a.w);
  u.z = pack2(b.x, b.y); u.w = pack2(b.z, b.w);
  *(U4*)(Vbf + (size_t)row * HW + p) = u;
}

// ================================================================ kernel 1
// Q/K projection: Q[p,o] = sum_c X[c,p] * W[o,c] + bias[o]   (bf16 out)
// Block: 256 thr (8 waves), tile = 128 p x 128 o, k-loop over c in steps of 32.
__global__ __launch_bounds__(256) void proj_kernel(
    const float* __restrict__ left, const float* __restrict__ right,
    const float* __restrict__ Wq, const float* __restrict__ bq,
    const float* __restrict__ Wk, const float* __restrict__ bk,
    bf16_t* __restrict__ Qbuf, bf16_t* __restrict__ Kbuf)
{
  __shared__ bf16_t As[128][40];   // [p_local][c_local] (transposed X tile)
  __shared__ bf16_t Ws[128][40];   // [o][c_local]

  const int t    = threadIdx.x;
  const int wv   = t >> 5;
  const int L    = t & 31;
  const int lrow = L & 15;
  const int kh   = L >> 4;

  const int pt    = blockIdx.x & 31;     // 32 p-tiles of 128
  const int rem   = blockIdx.x >> 5;     // 0..15
  const int which = rem & 1;             // 0 = Q-proj, 1 = K-proj
  const int sb    = rem >> 1;            // side*4 + b
  const int side  = sb >> 2, b = sb & 3;

  const int   tsel = which ? (1 - side) : side;
  const float* src = (tsel == 0) ? left : right;
  const float* W   = which ? Wk : Wq;
  const float* bias= which ? bk : bq;
  bf16_t* dst = (which ? Kbuf : Qbuf) + (size_t)sb * HW * HID;

  const int p0 = pt * 128;
  const float* srcb = src + (size_t)b * (2 * CH) * HW;

  v8f acc[8];
#pragma unroll
  for (int i = 0; i < 8; ++i)
#pragma unroll
    for (int k = 0; k < 8; ++k) acc[i][k] = 0.f;

  for (int cs = 0; cs < CH; cs += 32) {
    // stage X tile (32c x 128p), transposed into As[p][c]
#pragma unroll
    for (int i = 0; i < 4; ++i) {
      int id = t + 256 * i;
      int cl = id >> 5;              // 0..31
      int p4 = (id & 31) * 4;        // 0..124
      float4 x = *(const float4*)(srcb + (size_t)(cs + cl) * HW + p0 + p4);
      As[p4 + 0][cl] = (bf16_t)x.x;
      As[p4 + 1][cl] = (bf16_t)x.y;
      As[p4 + 2][cl] = (bf16_t)x.z;
      As[p4 + 3][cl] = (bf16_t)x.w;
    }
    // stage W tile (128o x 32c)
#pragma unroll
    for (int i = 0; i < 4; ++i) {
      int id = t + 256 * i;
      int o  = id >> 3;
      int c4 = (id & 7) * 4;
      float4 w4 = *(const float4*)(W + (size_t)o * CH + cs + c4);
      *(unsigned*)&Ws[o][c4 + 0] = pack2(w4.x, w4.y);
      *(unsigned*)&Ws[o][c4 + 2] = pack2(w4.z, w4.w);
    }
    __syncthreads();

    const int prow = wv * 16 + lrow;
    v16bf af = frag_ld(&As[prow][kh * 8], &As[prow][16 + kh * 8]);
#pragma unroll
    for (int ot = 0; ot < 8; ++ot) {
      v16bf bfrag = frag_ld(&Ws[ot * 16 + lrow][kh * 8],
                            &Ws[ot * 16 + lrow][16 + kh * 8]);
      acc[ot] = wmma_bf16(af, bfrag, acc[ot]);
    }
    __syncthreads();
  }

  // bias + store bf16 (D: lane = o column, VGPR r = row p (lanes>=16: +8))
#pragma unroll
  for (int ot = 0; ot < 8; ++ot) {
    int o = ot * 16 + lrow;
    float bv = bias[o];
#pragma unroll
    for (int r = 0; r < 8; ++r) {
      int p = p0 + wv * 16 + r + 8 * kh;
      dst[(size_t)p * HID + o] = (bf16_t)(acc[ot][r] + bv);
    }
  }
}

// ================================================================ kernel 2
// Flash attention: out[512+ch, q] = sum_p softmax_p(Q[q,:]·K[p,:]) * V[ch,p]
// Block: 256 thr = 8 waves = 2 q-subtiles(16) x 4 ch-groups(64). Key tile = 32.
// K/V tiles double-buffered and staged with global_load_async_to_lds_b128:
// tile kb+1 is in flight (ASYNCcnt=6/wave) while tile kb is consumed by WMMA.
__global__ __launch_bounds__(256) void attn_kernel(
    const bf16_t* __restrict__ Vbf,
    const bf16_t* __restrict__ Qbuf, const bf16_t* __restrict__ Kbuf,
    float* __restrict__ out)
{
  __shared__ bf16_t Ks[2][32][136];   // [buf][key][o]
  __shared__ bf16_t Vs[2][256][40];   // [buf][ch_local][key]
  __shared__ bf16_t Ps[8][16][40];    // per-wave P staging [q][key]
  __shared__ float  Xch[8][16];       // per-wave softmax-stat exchange

  const int t    = threadIdx.x;
  const int wv   = t >> 5;
  const int L    = t & 31;
  const int lrow = L & 15;
  const int kh   = L >> 4;

  const int idx  = blockIdx.x;
  const int cblk = idx & 1;
  const int qblk = (idx >> 1) & 127;
  const int sb   = idx >> 8;            // 0..7 = side*4 + b
  const int side = sb >> 2, b = sb & 3;

  const int vt = (side == 0) ? 1 : 0;   // V comes from the *other* tensor
  const bf16_t* vbase = Vbf + ((size_t)(vt * 4 + b) * CH + cblk * 256) * HW;
  const bf16_t* Qp    = Qbuf + (size_t)sb * HW * HID;
  const bf16_t* Kp    = Kbuf + (size_t)sb * HW * HID;

  const int qt     = wv & 1;
  const int cg     = wv >> 1;
  const int q0     = qblk * 32 + qt * 16;
  const int chloc0 = cg * 64;                 // within this block's 256 channels

  // Q fragments (A-layout: lane row = q, two contiguous 8-K chunks per half)
  v16bf qf[4];
  {
    const bf16_t* qrow = Qp + (size_t)(q0 + lrow) * HID;
#pragma unroll
    for (int f = 0; f < 4; ++f)
      qf[f] = frag_ld(qrow + f * 32 + kh * 8, qrow + f * 32 + 16 + kh * 8);
  }

  v8f o[4];
#pragma unroll
  for (int i = 0; i < 4; ++i)
#pragma unroll
    for (int k = 0; k < 8; ++k) o[i][k] = 0.f;
  float m[8], l[8];
#pragma unroll
  for (int r = 0; r < 8; ++r) { m[r] = -3.0e38f; l[r] = 0.f; }

  // async-issue one K/V tile (6 x b128 per thread -> ASYNCcnt += 6 per wave)
  auto issue_tile = [&](int kb) {
    const int pk = kb * 32;
    bf16_t (*ks)[136] = Ks[kb & 1];
    bf16_t (*vs)[40]  = Vs[kb & 1];
#pragma unroll
    for (int j = 0; j < 2; ++j) {         // K tile: 32 keys x 128 o = 8 KB
      int e = t * 2 + j;                  // 0..511
      async_b128(Kp + (size_t)pk * HID + e * 8, &ks[e >> 4][(e & 15) * 8]);
    }
#pragma unroll
    for (int j = 0; j < 4; ++j) {         // V tile: 256 ch x 32 keys = 16 KB
      int e  = t * 4 + j;                 // 0..1023
      int cl = e >> 2;
      int p8 = (e & 3) * 8;
      async_b128(vbase + (size_t)cl * HW + pk + p8, &vs[cl][p8]);
    }
  };

  issue_tile(0);

  const int NKB = HW / 32;
  for (int kb = 0; kb < NKB; ++kb) {
    bf16_t (*ks)[136] = Ks[kb & 1];
    bf16_t (*vs)[40]  = Vs[kb & 1];

    __syncthreads();            // all waves done reading buffer (kb+1)&1
    if (kb + 1 < NKB) {
      issue_tile(kb + 1);       // prefetch next tile into the other buffer
      WAIT_ASYNC(6);            // tile kb complete; kb+1 still in flight
    } else {
      WAIT_ASYNC(0);
    }
    __syncthreads();            // tile kb visible to every wave

    // S = Q x K^T : two 16x16 tiles (keys 0..15 / 16..31), contraction over o
    v8f s0, s1;
#pragma unroll
    for (int k = 0; k < 8; ++k) { s0[k] = 0.f; s1[k] = 0.f; }
#pragma unroll
    for (int f = 0; f < 4; ++f) {
      v16bf b0 = frag_ld(&ks[lrow][f * 32 + kh * 8],
                         &ks[lrow][f * 32 + 16 + kh * 8]);
      v16bf b1 = frag_ld(&ks[16 + lrow][f * 32 + kh * 8],
                         &ks[16 + lrow][f * 32 + 16 + kh * 8]);
      s0 = wmma_bf16(qf[f], b0, s0);
      s1 = wmma_bf16(qf[f], b1, s1);
    }

    // online softmax (rows = queries live in VGPRs; keys across 16-lane halves)
    float scl[8];
#pragma unroll
    for (int r = 0; r < 8; ++r) {
      float rm = fmaxf(s0[r], s1[r]);
#pragma unroll
      for (int off = 8; off; off >>= 1) rm = fmaxf(rm, __shfl_xor(rm, off, 32));
      float mn = fmaxf(m[r], rm);
      float e0 = __expf(s0[r] - mn);
      float e1 = __expf(s1[r] - mn);
      float rs = e0 + e1;
#pragma unroll
      for (int off = 8; off; off >>= 1) rs += __shfl_xor(rs, off, 32);
      scl[r] = __expf(m[r] - mn);
      l[r]   = l[r] * scl[r] + rs;
      m[r]   = mn;
      // stage P (bf16) as [q][key] for the P^T B-fragment of the 2nd GEMM
      int q = r + 8 * kh;
      Ps[wv][q][lrow]      = (bf16_t)e0;
      Ps[wv][q][16 + lrow] = (bf16_t)e1;
    }
    if (lrow == 0) {
#pragma unroll
      for (int r = 0; r < 8; ++r) Xch[wv][r + 8 * kh] = scl[r];
    }
    // Ps/Xch are wave-private; same-wave DS ops execute in order (DScnt),
    // only a compiler scheduling fence is needed -- no workgroup barrier.
    WAVE_FENCE();

    // O[ch][q] = O*scale[q] + V(16ch x 32p) x P^T(32p x 16q)
    float sq = Xch[wv][lrow];
    v16bf pf = frag_ld(&Ps[wv][lrow][kh * 8], &Ps[wv][lrow][16 + kh * 8]);
#pragma unroll
    for (int tt = 0; tt < 4; ++tt) {
#pragma unroll
      for (int r = 0; r < 8; ++r) o[tt][r] *= sq;
      int chl = chloc0 + tt * 16 + lrow;   // A-frag lane row = channel
      v16bf vf = frag_ld(&vs[chl][kh * 8], &vs[chl][16 + kh * 8]);
      o[tt] = wmma_bf16(vf, pf, o[tt]);
    }
  }

  // final 1/l and store: O-tile lane = q column -> contiguous stores along q
  if (lrow == 0) {
#pragma unroll
    for (int r = 0; r < 8; ++r) Xch[wv][r + 8 * kh] = l[r];
  }
  WAVE_FENCE();
  float inv = 1.0f / Xch[wv][lrow];

  float* ob = out + ((size_t)side * BB + b) * (2 * CH) * HW;
  int q = q0 + lrow;
#pragma unroll
  for (int tt = 0; tt < 4; ++tt)
#pragma unroll
    for (int r = 0; r < 8; ++r) {
      int ch = CH + cblk * 256 + chloc0 + tt * 16 + r + 8 * kh;
      ob[(size_t)ch * HW + q] = o[tt][r] * inv;
    }
}

// ================================================================ kernel 3
// homo term + q_orig copy: out[:, :512]=q_orig ; out[:, 512:] += qt*softmax_c(qo-qt)
__global__ __launch_bounds__(256) void homo_copy_kernel(
    const float* __restrict__ left, const float* __restrict__ right,
    float* __restrict__ out)
{
  const int idx  = blockIdx.x;
  const int sb   = idx >> 4;
  const int side = sb >> 2, b = sb & 3;
  const int p    = (idx & 15) * 256 + threadIdx.x;

  const float* src = ((side == 0) ? left : right) + (size_t)b * (2 * CH) * HW + p;
  float* ob = out + ((size_t)side * BB + b) * (2 * CH) * HW + p;

  float mx = -3.0e38f;
  for (int c = 0; c < CH; ++c) {
    float d = src[(size_t)c * HW] - src[(size_t)(CH + c) * HW];
    mx = fmaxf(mx, d);
  }
  float s = 0.f;
  for (int c = 0; c < CH; ++c) {
    float d = src[(size_t)c * HW] - src[(size_t)(CH + c) * HW];
    s += __expf(d - mx);
  }
  float invs = 1.0f / s;
  for (int c = 0; c < CH; ++c) {
    float qo = src[(size_t)c * HW];
    float qt = src[(size_t)(CH + c) * HW];
    float w  = __expf(qo - qt - mx) * invs;
    ob[(size_t)c * HW]        = qo;
    ob[(size_t)(CH + c) * HW] += qt * w;
  }
}

// ================================================================ launch
extern "C" void kernel_launch(void* const* d_in, const int* in_sizes, int n_in,
                              void* d_out, int out_size, void* d_ws, size_t ws_size,
                              hipStream_t stream) {
  const float* left  = (const float*)d_in[0];
  const float* right = (const float*)d_in[1];
  const float* Wq    = (const float*)d_in[2];
  const float* bq    = (const float*)d_in[3];
  const float* Wk    = (const float*)d_in[4];
  const float* bk    = (const float*)d_in[5];
  float* out = (float*)d_out;

  bf16_t* Qbuf = (bf16_t*)d_ws;                     // 8 * 4096 * 128 bf16
  bf16_t* Kbuf = Qbuf + (size_t)8 * HW * HID;       // + 8.4 MB
  bf16_t* Vbf  = Kbuf + (size_t)8 * HW * HID;       // + 33.5 MB

  vconv_kernel<<<8192, 256, 0, stream>>>(left, right, Vbf);
  proj_kernel<<<512, 256, 0, stream>>>(left, right, Wq, bq, Wk, bk, Qbuf, Kbuf);
  attn_kernel<<<2048, 256, 0, stream>>>(Vbf, Qbuf, Kbuf, out);
  homo_copy_kernel<<<128, 256, 0, stream>>>(left, right, out);
}
